// MeanPooledSpanEmbeddingLayer_40389872451847
// MI455X (gfx1250) — compile-verified, matrix operand order checked by reading, not run on hardware
//
#include <hip/hip_runtime.h>
#include <hip/hip_bf16.h>
#include <math.h>

// ---------------------------------------------------------------------------
// MeanPooledSpanEmbeddingLayer for MI455X (gfx1250, wave32, WMMA)
// GEMMs via v_wmma_f32_16x16x32_bf16. Weights are transpose-converted to
// bf16 [N,K] so both A and B fragments load as two contiguous b128 chunks
// (ISA 16-bit fragment layout: K = kb + 8*half + {0..7} and +16).
// The K-loop is software-pipelined (two-stage ping-pong): next stage's
// b128 loads issue before current stage's 4 WMMAs, so loads overlap math.
// Workload is HBM-bound (~565 GFLOP vs ~1.3 GB @ 23.3 TB/s) -> fuse all
// epilogues; bf16 weights (~42 MB) stay resident in the 192 MB L2.
// ---------------------------------------------------------------------------

typedef __attribute__((ext_vector_type(16))) __bf16 v16bf;
typedef __attribute__((ext_vector_type(8)))  __bf16 bf16x8;
typedef __attribute__((ext_vector_type(8)))  float  v8f;

constexpr int H_DIM  = 2048;
constexpr int B_SZ   = 4;
constexpr int S_LEN  = 4096;
constexpr int SPANS  = 512;                 // spans per batch item
constexpr int M_TOK  = B_SZ * S_LEN;        // 16384 token rows
constexpr int N_SPAN = B_SZ * SPANS;        // 2048 spans
constexpr float LN_EPS = 1e-5f;

enum { MODE_PLAIN = 0, MODE_GELU = 1, MODE_SIGMOID = 2 };

// ---------------------------------------------------------------------------
// Transpose-convert weight W[K,N] f32 -> BT[N,K] bf16 (32x32 LDS tile).
// ---------------------------------------------------------------------------
__global__ __launch_bounds__(256)
void cvt_transpose_bf16(const float* __restrict__ W, __bf16* __restrict__ BT,
                        int K, int N) {
  __shared__ float tile[32][33];
  const int k0 = blockIdx.y * 32, n0 = blockIdx.x * 32;
  const int tx = threadIdx.x & 31, ty = threadIdx.x >> 5;   // 8 rows/pass
#pragma unroll
  for (int r = ty; r < 32; r += 8)
    tile[r][tx] = W[(size_t)(k0 + r) * N + (n0 + tx)];
  __syncthreads();
#pragma unroll
  for (int r = ty; r < 32; r += 8)
    BT[(size_t)(n0 + r) * K + (k0 + tx)] = (__bf16)tile[tx][r];
}

// ---------------------------------------------------------------------------
// Token embedding gather: f32 + bf16 shadow, float4-vectorized.
// ---------------------------------------------------------------------------
__global__ __launch_bounds__(256)
void gather_kernel(const long long* __restrict__ ids,
                   const float* __restrict__ table,
                   float* __restrict__ Xf, __bf16* __restrict__ Xb) {
  const int t = blockIdx.x;
  const float4* src = (const float4*)(table + (size_t)ids[t] * H_DIM);
  float4* dstf = (float4*)(Xf + (size_t)t * H_DIM);
  __bf16* dstb = Xb + (size_t)t * H_DIM;
  for (int h4 = threadIdx.x; h4 < H_DIM / 4; h4 += blockDim.x) {
    float4 v = src[h4];
    dstf[h4] = v;
    dstb[h4 * 4 + 0] = (__bf16)v.x;
    dstb[h4 * 4 + 1] = (__bf16)v.y;
    dstb[h4 * 4 + 2] = (__bf16)v.z;
    dstb[h4 * 4 + 3] = (__bf16)v.w;
  }
}

// ---------------------------------------------------------------------------
// Fused WMMA GEMM:  C = epilogue( [A0|A1] @ W + bias (+resid) )
//   A0: [Mr,K0] bf16 row-major, A1 (optional): [Mr,K1] bf16 row-major
//   BT: [N, K0+K1] bf16 (transposed weight) -> fragments = 2x b128 loads
// Block = 128 threads = 4 waves; each wave computes a 16x64 strip with
// 4 accumulators, reusing one A fragment across 4 WMMAs per K-step.
// Requires (K0+K1)/32 even (K = 2048 or 4096 here). Grid: (N/64, Mr/64).
// ---------------------------------------------------------------------------
__global__ __launch_bounds__(128)
void gemm_wmma_bf16(const __bf16* __restrict__ A0, const __bf16* __restrict__ A1,
                    const __bf16* __restrict__ BT,
                    const float* __restrict__ bias,
                    const float* __restrict__ resid,
                    const float* __restrict__ rowscale,
                    float* __restrict__ Cf, __bf16* __restrict__ Cb,
                    int Mr, int N, int K0, int K1, int mode) {
  const int wave = threadIdx.x >> 5;
  const int lane = threadIdx.x & 31;
  const int hl   = lane >> 4;            // lane half selects K sub-block
  const int idx  = lane & 15;
  const int row0 = blockIdx.y * 64 + wave * 16;
  const int col0 = blockIdx.x * 64;
  const int K    = K0 + K1;
  const int row  = row0 + idx;           // A rows: lanes 0-15 == lanes 16-31

  union Frag { bf16x8 h[2]; v16bf v; };
  v8f acc[4] = {};

  auto load_stage = [&](Frag& a, Frag* b, int kb) {
    // uniform split: each 32-wide K step lies entirely in A0 or A1
    const __bf16* arow = (kb < K0)
        ? (A0 + (size_t)row * K0 + kb)
        : (A1 + (size_t)row * K1 + (kb - K0));
    a.h[0] = *(const bf16x8*)(arow + hl * 8);        // K = kb+8h+0..7
    a.h[1] = *(const bf16x8*)(arow + 16 + hl * 8);   // K = kb+16+8h+0..7
#pragma unroll
    for (int t = 0; t < 4; ++t) {
      const __bf16* brow = BT + (size_t)(col0 + t * 16 + idx) * K + kb;
      b[t].h[0] = *(const bf16x8*)(brow + hl * 8);
      b[t].h[1] = *(const bf16x8*)(brow + 16 + hl * 8);
    }
  };
  auto mma_stage = [&](Frag& a, Frag* b) {
#pragma unroll
    for (int t = 0; t < 4; ++t)
      acc[t] = __builtin_amdgcn_wmma_f32_16x16x32_bf16(
          false, a.v, false, b[t].v, (short)0, acc[t], false, false);
  };

  // two-stage ping-pong software pipeline (K/32 is even)
  Frag aA, bA[4], aB, bB[4];
  load_stage(aA, bA, 0);
  for (int kb = 32; kb < K; kb += 64) {
    load_stage(aB, bB, kb);              // stage B loads overlap stage A math
    mma_stage(aA, bA);
    if (kb + 32 < K) {
      load_stage(aA, bA, kb + 32);       // stage A loads overlap stage B math
      __builtin_prefetch(BT + (size_t)(col0 + idx) * K + kb + 64, 0, 1);
    }
    mma_stage(aB, bB);
  }

  union CF { float e[8]; v8f v; } c;
#pragma unroll
  for (int t = 0; t < 4; ++t) {
    c.v = acc[t];
    const int n = col0 + t * 16 + idx;
    const float bn = bias[n];
#pragma unroll
    for (int r = 0; r < 8; ++r) {
      const int m = row0 + hl * 8 + r;   // D layout: M = vgpr + 8*half
      float val = c.e[r] + bn;
      if (resid) val += resid[(size_t)m * N + n];
      if (mode == MODE_GELU) {
        val = 0.5f * val * (1.0f + erff(val * 0.70710678118654752f));
      } else if (mode == MODE_SIGMOID) {
        val = 1.0f / (1.0f + __expf(-val));
      }
      if (rowscale) val *= rowscale[m];
      if (Cf) Cf[(size_t)m * N + n] = val;
      if (Cb) Cb[(size_t)m * N + n] = (__bf16)val;
    }
  }
}

// ---------------------------------------------------------------------------
// Ragged span mean-pool. One block per span; span_ids row (16 KB) in LDS,
// single sweep over S tokens. Emits bf16 (pooled + type_emb) + counts.
// ---------------------------------------------------------------------------
__global__ __launch_bounds__(256)
void pool_kernel(const float* __restrict__ X2f,
                 const int* __restrict__ span_ids,
                 const int* __restrict__ span_type_ids,
                 const float* __restrict__ type_table,
                 __bf16* __restrict__ Pb, float* __restrict__ cnt) {
  __shared__ int sid[S_LEN];
  const int s     = blockIdx.x;
  const int b     = s / SPANS;
  const int local = s % SPANS;
  const int tid   = threadIdx.x;

  for (int i = tid; i < S_LEN; i += 256) sid[i] = span_ids[b * S_LEN + i];
  __syncthreads();

  float acc[8] = {0.f, 0.f, 0.f, 0.f, 0.f, 0.f, 0.f, 0.f};
  int count = 0;
  const float* base = X2f + (size_t)b * S_LEN * H_DIM;
  for (int t = 0; t < S_LEN; ++t) {
    if (sid[t] == local) {               // block-uniform branch
      ++count;
      const float* rowp = base + (size_t)t * H_DIM;
#pragma unroll
      for (int j = 0; j < 8; ++j) acc[j] += rowp[tid + j * 256];
    }
  }
  const float inv = 1.0f / fmaxf((float)count, 1.0f);
  const int  ty   = span_type_ids[s];
#pragma unroll
  for (int j = 0; j < 8; ++j) {
    const int h = tid + j * 256;
    Pb[(size_t)s * H_DIM + h] =
        (__bf16)(acc[j] * inv + type_table[(size_t)ty * H_DIM + h]);
  }
  if (tid == 0) cnt[s] = (float)count;
}

// ---------------------------------------------------------------------------
// Per-token span select + LayerNorm + span mask.
// ---------------------------------------------------------------------------
__global__ __launch_bounds__(256)
void select_ln_kernel(const float* __restrict__ X2f, const float* __restrict__ Ef,
                      const int* __restrict__ span_ids, const float* __restrict__ cnt,
                      const float* __restrict__ ln_g, const float* __restrict__ ln_b,
                      float* __restrict__ SEf, __bf16* __restrict__ SNb,
                      float* __restrict__ mask_out) {
  __shared__ float red[256];
  const int t   = blockIdx.x;
  const int tid = threadIdx.x;
  const int seg = span_ids[t] + (t / S_LEN) * SPANS;
  const bool valid = cnt[seg] > 1.0f;
  const float* src = valid ? (Ef + (size_t)seg * H_DIM)
                           : (X2f + (size_t)t * H_DIM);
  float x[8];
  float sum = 0.f;
#pragma unroll
  for (int j = 0; j < 8; ++j) {
    const int h = tid + j * 256;
    x[j] = src[h];
    SEf[(size_t)t * H_DIM + h] = x[j];
    sum += x[j];
  }
  red[tid] = sum; __syncthreads();
  for (int w = 128; w > 0; w >>= 1) { if (tid < w) red[tid] += red[tid + w]; __syncthreads(); }
  const float mean = red[0] / (float)H_DIM;
  __syncthreads();
  float vs = 0.f;
#pragma unroll
  for (int j = 0; j < 8; ++j) { const float d = x[j] - mean; vs += d * d; }
  red[tid] = vs; __syncthreads();
  for (int w = 128; w > 0; w >>= 1) { if (tid < w) red[tid] += red[tid + w]; __syncthreads(); }
  const float rstd = rsqrtf(red[0] / (float)H_DIM + LN_EPS);
#pragma unroll
  for (int j = 0; j < 8; ++j) {
    const int h = tid + j * 256;
    SNb[(size_t)t * H_DIM + h] =
        (__bf16)((x[j] - mean) * rstd * ln_g[h] + ln_b[h]);
  }
  if (tid == 0) mask_out[t] = valid ? 1.0f : 0.0f;
}

// ---------------------------------------------------------------------------
// combined = (1 - gate) * tok + gate * span_emb  (gate already mask-scaled)
// ---------------------------------------------------------------------------
__global__ __launch_bounds__(256)
void combine_kernel(const float* __restrict__ X2f,
                    const float* __restrict__ SEf,
                    const float* __restrict__ gate,
                    float* __restrict__ out, long n4) {
  long i = (long)blockIdx.x * blockDim.x + threadIdx.x;
  if (i < n4) {
    float4 g = ((const float4*)gate)[i];
    float4 a = ((const float4*)X2f)[i];
    float4 s = ((const float4*)SEf)[i];
    float4 o;
    o.x = (1.0f - g.x) * a.x + g.x * s.x;
    o.y = (1.0f - g.y) * a.y + g.y * s.y;
    o.z = (1.0f - g.z) * a.z + g.z * s.z;
    o.w = (1.0f - g.w) * a.w + g.w * s.w;
    ((float4*)out)[i] = o;
  }
}

// ---------------------------------------------------------------------------
extern "C" void kernel_launch(void* const* d_in, const int* in_sizes, int n_in,
                              void* d_out, int out_size, void* d_ws, size_t ws_size,
                              hipStream_t stream) {
  (void)in_sizes; (void)n_in; (void)out_size; (void)ws_size;

  const long long* input_ids     = (const long long*)d_in[0];   // int64 [B,S]
  const int*   span_ids          = (const int*)d_in[1];         // int32 [B,S]
  const int*   span_type_ids     = (const int*)d_in[2];         // int32 [B*SPANS]
  const float* tok_table         = (const float*)d_in[3];
  const float* W1                = (const float*)d_in[4];
  const float* b1                = (const float*)d_in[5];
  const float* W2                = (const float*)d_in[6];
  const float* b2                = (const float*)d_in[7];
  const float* type_table        = (const float*)d_in[8];
  const float* Wp                = (const float*)d_in[9];
  const float* bp                = (const float*)d_in[10];
  const float* ln_g              = (const float*)d_in[11];
  const float* ln_b              = (const float*)d_in[12];
  const float* Wg                = (const float*)d_in[13];
  const float* bg                = (const float*)d_in[14];

  const size_t MH = (size_t)M_TOK * H_DIM;
  float* out_combined = (float*)d_out;            // [B,S,H]
  float* out_gate     = out_combined + MH;        // [B,S,H]
  float* out_mask     = out_gate + MH;            // [B,S]

  // --- bump allocator over workspace ---
  char* wsp = (char*)d_ws;
  auto alloc = [&](size_t bytes) -> void* {
    void* p = wsp;
    wsp += (bytes + 255) & ~(size_t)255;
    return p;
  };
  float*  Xf  = (float*) alloc(MH * sizeof(float));    // gathered tok (f32)
  __bf16* Xb  = (__bf16*)alloc(MH * sizeof(__bf16));   // gathered tok (bf16)
  __bf16* H1b = (__bf16*)alloc(MH * sizeof(__bf16));   // gelu(X@W1+b1)
  float*  X2f = (float*) alloc(MH * sizeof(float));    // tok after adapter
  __bf16* X2b = (__bf16*)alloc(MH * sizeof(__bf16));
  float*  SEf = (float*) alloc(MH * sizeof(float));    // span_emb per token
  __bf16* SNb = (__bf16*)alloc(MH * sizeof(__bf16));   // LN(span_emb)
  const size_t HH = (size_t)H_DIM * H_DIM;
  __bf16* W1t = (__bf16*)alloc(HH * sizeof(__bf16));   // [N,K] bf16
  __bf16* W2t = (__bf16*)alloc(HH * sizeof(__bf16));
  __bf16* Wpt = (__bf16*)alloc(HH * sizeof(__bf16));
  __bf16* Wgt = (__bf16*)alloc(2 * HH * sizeof(__bf16));
  __bf16* Pb  = (__bf16*)alloc((size_t)N_SPAN * H_DIM * sizeof(__bf16));
  float*  Ef  = (float*) alloc((size_t)N_SPAN * H_DIM * sizeof(float));
  float*  cnt = (float*) alloc((size_t)N_SPAN * sizeof(float));

  // --- weight transpose-conversion to bf16 [N,K] ---
  {
    dim3 gHH(H_DIM / 32, H_DIM / 32);
    cvt_transpose_bf16<<<gHH, 256, 0, stream>>>(W1, W1t, H_DIM, H_DIM);
    cvt_transpose_bf16<<<gHH, 256, 0, stream>>>(W2, W2t, H_DIM, H_DIM);
    cvt_transpose_bf16<<<gHH, 256, 0, stream>>>(Wp, Wpt, H_DIM, H_DIM);
    dim3 gWg(H_DIM / 32, 2 * H_DIM / 32);
    cvt_transpose_bf16<<<gWg, 256, 0, stream>>>(Wg, Wgt, 2 * H_DIM, H_DIM);
  }

  // --- token gather ---
  gather_kernel<<<M_TOK, 256, 0, stream>>>(input_ids, tok_table, Xf, Xb);

  // --- adapter GEMM 1: H1 = gelu(X @ W1 + b1)  [bf16 out] ---
  {
    dim3 grid(H_DIM / 64, M_TOK / 64);
    gemm_wmma_bf16<<<grid, 128, 0, stream>>>(
        Xb, nullptr, W1t, b1, nullptr, nullptr,
        nullptr, H1b, M_TOK, H_DIM, H_DIM, 0, MODE_GELU);
  }
  // --- adapter GEMM 2: X2 = X + H1 @ W2 + b2  [f32 + bf16 out] ---
  {
    dim3 grid(H_DIM / 64, M_TOK / 64);
    gemm_wmma_bf16<<<grid, 128, 0, stream>>>(
        H1b, nullptr, W2t, b2, Xf, nullptr,
        X2f, X2b, M_TOK, H_DIM, H_DIM, 0, MODE_PLAIN);
  }

  // --- ragged span mean pool (+ type embedding) ---
  pool_kernel<<<N_SPAN, 256, 0, stream>>>(X2f, span_ids, span_type_ids,
                                          type_table, Pb, cnt);

  // --- span projection GEMM: E = P @ Wp + bp  [f32 out] ---
  {
    dim3 grid(H_DIM / 64, N_SPAN / 64);
    gemm_wmma_bf16<<<grid, 128, 0, stream>>>(
        Pb, nullptr, Wpt, bp, nullptr, nullptr,
        Ef, nullptr, N_SPAN, H_DIM, H_DIM, 0, MODE_PLAIN);
  }

  // --- per-token select + LayerNorm + mask ---
  select_ln_kernel<<<M_TOK, 256, 0, stream>>>(X2f, Ef, span_ids, cnt,
                                              ln_g, ln_b, SEf, SNb, out_mask);

  // --- gate GEMM: gate = sigmoid([X2 | SN] @ Wg + bg) * mask[row] ---
  {
    dim3 grid(H_DIM / 64, M_TOK / 64);
    gemm_wmma_bf16<<<grid, 128, 0, stream>>>(
        X2b, SNb, Wgt, bg, nullptr, out_mask,
        out_gate, nullptr, M_TOK, H_DIM, H_DIM, H_DIM, MODE_SIGMOID);
  }

  // --- final combine (float4) ---
  {
    long n4 = (long)(MH / 4);
    combine_kernel<<<(unsigned)((n4 + 255) / 256), 256, 0, stream>>>(
        X2f, SEf, out_gate, out_combined, n4);
  }
}
